// ConvVMamba_31275951849887
// MI455X (gfx1250) — matrix operand order repaired and checked by reading, not compiled, not run on hardware
//
#include <hip/hip_runtime.h>
#include <hip/hip_bf16.h>
#include <math.h>

// ---------------- problem constants ----------------
#define BB   8
#define CC   128
#define HHs  64
#define WWs  64
#define LLs  4096          // H*W
#define CINs 192
#define NSt  4             // d_state
#define KDr  4             // scan directions
#define RDt  8             // dt_rank
#define DFFs 512
#define EPSF 1e-5f

typedef __attribute__((ext_vector_type(16))) _Float16     v16h;
typedef __attribute__((ext_vector_type(8)))  float        v8f;
typedef __attribute__((ext_vector_type(4)))  unsigned int v4u;
typedef __attribute__((ext_vector_type(8)))  int          v8i;
typedef __attribute__((ext_vector_type(4)))  int          v4i;

#if defined(__has_builtin)
#if __has_builtin(__builtin_amdgcn_tensor_load_to_lds) && \
    __has_builtin(__builtin_amdgcn_s_wait_tensorcnt)
#define VM_TDM 1
#endif
#endif

__device__ __forceinline__ float vm_sigmoid(float x) { return 1.0f / (1.0f + expf(-x)); }
__device__ __forceinline__ float vm_softplus(float x) { return (x > 20.0f) ? x : log1pf(expf(x)); }
__device__ __forceinline__ float vm_gelu(float x) { return 0.5f * x * (1.0f + erff(x * 0.70710678118654752f)); }

// ---------------- TDM 2-D tile load (32 rows x 128 cols of f32, row stride = LLs) ----------------
#ifdef VM_TDM
__device__ __forceinline__ void vm_tdm_load_tile_f32(const float* gsrc, void* ldsDst)
{
    unsigned long long ga = (unsigned long long)(const void*)gsrc;
    unsigned long long la = (unsigned long long)ldsDst;      // low 32 bits = LDS byte address
    const unsigned tileCols = 128u, tileRows = 32u;
    const unsigned long long rowStride = (unsigned long long)LLs;  // in elements (data_size units)
    const unsigned tensorCols = (unsigned)LLs, tensorRows = 32u;

    v4u g0;
    g0[0] = 1u;                                              // count = 1 valid descriptor
    g0[1] = (unsigned)la;                                    // lds_addr
    g0[2] = (unsigned)(ga & 0xffffffffull);                  // global_addr[31:0]
    g0[3] = (unsigned)((ga >> 32) & 0x01ffffffull) | (2u << 30); // addr[56:32] | type=2

    v8i g1;
    g1[0] = (int)(2u << 16);                                  // data_size = 2 -> 4 bytes
    g1[1] = (int)((tensorCols & 0xffffu) << 16);              // tensor_dim0[15:0] @ bits 63:48
    g1[2] = (int)((tensorCols >> 16) & 0xffffu) |             // tensor_dim0[31:16]
            (int)((tensorRows & 0xffffu) << 16);              // tensor_dim1[15:0]
    g1[3] = (int)((tensorRows >> 16) & 0xffffu) |             // tensor_dim1[31:16]
            (int)((tileCols & 0xffffu) << 16);                // tile_dim0
    g1[4] = (int)(tileRows & 0xffffu);                        // tile_dim1 (tile_dim2 = 0)
    g1[5] = (int)(rowStride & 0xffffffffull);                 // tensor_dim0_stride[31:0]
    g1[6] = (int)((rowStride >> 32) & 0xffffull);             // stride[47:32] (dim1_stride = 0)
    g1[7] = 0;

    v4i gz; gz[0] = 0; gz[1] = 0; gz[2] = 0; gz[3] = 0;
#if defined(__clang_major__) && (__clang_major__ >= 23)
    v8i gz8; gz8[0]=0; gz8[1]=0; gz8[2]=0; gz8[3]=0; gz8[4]=0; gz8[5]=0; gz8[6]=0; gz8[7]=0;
    __builtin_amdgcn_tensor_load_to_lds(g0, g1, gz, gz, gz8, 0);
#else
    __builtin_amdgcn_tensor_load_to_lds(g0, g1, gz, gz, 0);
#endif
}
#endif

// ---------------- generic channel-first WMMA GEMM (double-buffered TDM pipeline) ----------------
// Per "bk" slab:  Y[M,L] (+res) = act( W[M,K] * X[K,L] + bias )
// X is split into two sources (concat along K); tiles: M0=128 rows, N=128 cols, Kstep=32.
// Pipeline: stage tile i+1 (A via VALU, B via Tensor Data Mover), s_wait_tensorcnt 1
// (TENSORcnt retires in order -> tile i resident), barrier, 8 WMMAs on tile i.
template <int ACT>  // 0 = none, 1 = exact GELU
__global__ __launch_bounds__(256)
void vm_gemm_cf(const float* __restrict__ Wm, int wMod, int wStride,
                const float* __restrict__ X1, int K1, int xbs1,
                const float* __restrict__ X2, int K2, int xbs2,
                const float* __restrict__ bias,
                const float* __restrict__ res,
                float* __restrict__ Y, int M)
{
    constexpr int LT = LLs / 128;
    const int bk   = blockIdx.x / LT;
    const int lt   = blockIdx.x % LT;
    const int m0   = blockIdx.y * 128;
    const int tid  = threadIdx.x;
    const int wid  = tid >> 5;
    const int lane = tid & 31;
    const int K    = K1 + K2;
    const int NK   = K >> 5;
    const float* Wb = Wm + (wMod > 1 ? (bk % wMod) * wStride : 0);

    __shared__ _Float16 sA[2][128 * 32];   // weights tile (f16), double buffered
    __shared__ float    sB[2][32 * 128];   // activation tile (f32, TDM target), double buffered

    v8f acc[8];
#pragma unroll
    for (int nt = 0; nt < 8; ++nt)
#pragma unroll
        for (int j = 0; j < 8; ++j) acc[nt][j] = 0.0f;

    const int mrow = (wid << 4) + (lane & 15);
    const int hi   = lane >> 4;

    // ---- tile stager (A: cooperative f32->f16; B: one TDM descriptor from wave 0) ----
    auto stage = [&](int ti, int buf) {
        const int k0 = ti << 5;
#pragma unroll 2
        for (int i = tid; i < 128 * 32; i += 256) {
            int r = i >> 5, kk = i & 31;
            int row = m0 + r; if (row >= M) row = M - 1;   // clamp (store guarded later)
            sA[buf][i] = (_Float16)Wb[row * K + k0 + kk];
        }
        const float* src; int rr0;
        if (k0 < K1) { src = X1 + (long long)bk * xbs1; rr0 = k0; }
        else         { src = X2 + (long long)bk * xbs2; rr0 = k0 - K1; }
        const float* tileSrc = src + (long long)rr0 * LLs + lt * 128;
#ifdef VM_TDM
        if (wid == 0) vm_tdm_load_tile_f32(tileSrc, (void*)&sB[buf][0]);
#else
#pragma unroll 2
        for (int i = tid; i < 32 * 128; i += 256) {
            int kk = i >> 7, col = i & 127;
            sB[buf][i] = tileSrc[kk * LLs + col];
        }
#endif
    };

    // ---- prologue: stage tile 0 ----
    stage(0, 0);

    for (int i = 0; i < NK; ++i) {
        const int cur = i & 1;
        if (i + 1 < NK) {
            stage(i + 1, cur ^ 1);           // overlap next tile's DMA with this tile's math
#ifdef VM_TDM
            if (wid == 0) __builtin_amdgcn_s_wait_tensorcnt(1);  // tile i resident (in-order)
#endif
        } else {
#ifdef VM_TDM
            if (wid == 0) __builtin_amdgcn_s_wait_tensorcnt(0);
#endif
        }
        __syncthreads();

        // --- A fragment: 16x32 f16 (row = lane&15, K split by lane>>4) ---
        v16h afr;
#pragma unroll
        for (int v = 0; v < 8; ++v) {
            int kk = ((v & 3) << 1) + (hi << 3) + ((v >> 2) << 4);
            afr[2 * v]     = sA[cur][mrow * 32 + kk];
            afr[2 * v + 1] = sA[cur][mrow * 32 + kk + 1];
        }
        // --- 8 N-tiles of B fragment + WMMA ---
#pragma unroll
        for (int nt = 0; nt < 8; ++nt) {
            v16h bfr;
            const int colb = (nt << 4) + (lane & 15);
#pragma unroll
            for (int v = 0; v < 8; ++v) {
                int kk = ((v & 3) << 1) + (hi << 3) + ((v >> 2) << 4);
                bfr[2 * v]     = (_Float16)sB[cur][kk * 128 + colb];
                bfr[2 * v + 1] = (_Float16)sB[cur][(kk + 1) * 128 + colb];
            }
            acc[nt] = __builtin_amdgcn_wmma_f32_16x16x32_f16(
                false, afr, false, bfr, (short)0, acc[nt], false, false);
        }
        __syncthreads();   // compute(i) done before tile i+2 overwrites these buffers
    }

    // --- epilogue: bias -> activation -> residual -> store ---
    const long long ybase = (long long)bk * M * LLs;
#pragma unroll
    for (int nt = 0; nt < 8; ++nt) {
        const int col = lt * 128 + (nt << 4) + (lane & 15);
#pragma unroll
        for (int r = 0; r < 8; ++r) {
            const int row = m0 + (wid << 4) + r + (hi << 3);
            if (row < M) {
                float v = acc[nt][r];
                if (bias) v += bias[row];
                if (ACT == 1) v = vm_gelu(v);
                if (res) v += res[ybase + (long long)row * LLs + col];
                Y[ybase + (long long)row * LLs + col] = v;
            }
        }
    }
}

// ---------------- LayerNorm over channel (channel-first) ----------------
__global__ __launch_bounds__(256)
void vm_ln_cf(const float* __restrict__ in, const float* __restrict__ g,
              const float* __restrict__ be, float* __restrict__ out)
{
    const int idx = blockIdx.x * 256 + threadIdx.x;   // over B*L
    const int b = idx / LLs, l = idx % LLs;
    const float* p = in + (long long)b * CC * LLs + l;
    float s = 0.f, s2 = 0.f;
#pragma unroll 4
    for (int c = 0; c < CC; ++c) { float v = p[c * LLs]; s += v; s2 += v * v; }
    const float mu = s * (1.0f / CC);
    const float var = s2 * (1.0f / CC) - mu * mu;
    const float rs = rsqrtf(var + EPSF);
    float* q = out + (long long)b * CC * LLs + l;
#pragma unroll 4
    for (int c = 0; c < CC; ++c) q[c * LLs] = (p[c * LLs] - mu) * rs * g[c] + be[c];
}

// ---------------- depthwise 3x3 conv + bias + SiLU ----------------
__global__ __launch_bounds__(256)
void vm_dwconv_silu(const float* __restrict__ in, const float* __restrict__ w,
                    const float* __restrict__ bias, float* __restrict__ out)
{
    const int idx = blockIdx.x * 256 + threadIdx.x;   // over B*C*L
    const int l = idx % LLs, c = (idx / LLs) % CC;
    const int h = l / WWs, wv = l % WWs;
    const float* ip = in + (long long)(idx / LLs) * LLs;
    const float* wp = w + c * 9;
    float a = bias[c];
#pragma unroll
    for (int i = 0; i < 3; ++i)
#pragma unroll
        for (int j = 0; j < 3; ++j) {
            int h2 = h + i - 1, w2 = wv + j - 1;
            if (h2 >= 0 && h2 < HHs && w2 >= 0 && w2 < WWs)
                a += wp[i * 3 + j] * ip[h2 * WWs + w2];
        }
    out[idx] = a * vm_sigmoid(a);
}

// ---------------- build the 4 scan orderings xs[B,K,C,L] ----------------
__global__ __launch_bounds__(256)
void vm_build_xs(const float* __restrict__ u, float* __restrict__ xs)
{
    const int idx = blockIdx.x * 256 + threadIdx.x;   // over B*C*L
    const int l = idx % LLs, c = (idx / LLs) % CC, b = idx / (CC * LLs);
    const float v = u[idx];
    const int h = l / WWs, w = l % WWs;
    const int lt = w * HHs + h;                        // transposed position
    const long long slab = (long long)CC * LLs;
    float* base = xs + (long long)b * KDr * slab + (long long)c * LLs;
    base[0 * slab + l]             = v;
    base[1 * slab + lt]            = v;
    base[2 * slab + (LLs - 1 - l)] = v;
    base[3 * slab + (LLs - 1 - lt)] = v;
}

// ---------------- selective scan (block-parallel linear recurrence) ----------------
// One block per (b,k,c); 256 threads x 16 timesteps; N=4 states per thread.
// Output y is atomically merged (cross-merge permutation folded in).
__global__ __launch_bounds__(256)
void vm_scan(const float* __restrict__ xs, const float* __restrict__ xdbl,
             const float* __restrict__ dtw, const float* __restrict__ dtb,
             const float* __restrict__ Alogs, const float* __restrict__ Ds,
             float* __restrict__ ymerge)
{
    const int bkc = blockIdx.x;
    const int c = bkc % CC;
    const int k = (bkc / CC) % KDr;
    const int b = bkc / (CC * KDr);
    const int tid = threadIdx.x;
    constexpr int T = LLs / 256;      // 16
    const int l0 = tid * T;

    const float* xd = xdbl + (long long)(b * KDr + k) * (RDt + 2 * NSt) * LLs;
    const float* xr = xs + ((long long)(b * KDr + k) * CC + c) * LLs;
    const float* dw = dtw + (k * CC + c) * RDt;
    const float dbv = dtb[k * CC + c];
    const float Dv = Ds[k * CC + c];
    float An[NSt];
#pragma unroll
    for (int n = 0; n < NSt; ++n) An[n] = -expf(Alogs[(k * CC + c) * NSt + n]);

    __shared__ float sP[NSt * 256];
    __shared__ float sQ[NSt * 256];

    // pass 1: per-chunk composed transform (P, Q): h_out = P*h_in + Q
    float P[NSt], Q[NSt];
#pragma unroll
    for (int n = 0; n < NSt; ++n) { P[n] = 1.0f; Q[n] = 0.0f; }
    for (int t = 0; t < T; ++t) {
        const int l = l0 + t;
        float d = dbv;
#pragma unroll
        for (int r = 0; r < RDt; ++r) d += dw[r] * xd[r * LLs + l];
        d = vm_softplus(d);
        const float xv = xr[l];
#pragma unroll
        for (int n = 0; n < NSt; ++n) {
            const float a = expf(d * An[n]);
            Q[n] = a * Q[n] + d * xv * xd[(RDt + n) * LLs + l];
            P[n] *= a;
        }
    }
#pragma unroll
    for (int n = 0; n < NSt; ++n) { sP[n * 256 + tid] = P[n]; sQ[n * 256 + tid] = Q[n]; }
    __syncthreads();

    // Hillis-Steele inclusive scan of affine transforms
    for (int off = 1; off < 256; off <<= 1) {
        float p1[NSt], q1[NSt], p2[NSt], q2[NSt];
        const bool has = (tid >= off);
#pragma unroll
        for (int n = 0; n < NSt; ++n) { p2[n] = sP[n * 256 + tid]; q2[n] = sQ[n * 256 + tid]; }
        if (has)
#pragma unroll
            for (int n = 0; n < NSt; ++n) { p1[n] = sP[n * 256 + tid - off]; q1[n] = sQ[n * 256 + tid - off]; }
        __syncthreads();
        if (has)
#pragma unroll
            for (int n = 0; n < NSt; ++n) {
                sP[n * 256 + tid] = p1[n] * p2[n];
                sQ[n * 256 + tid] = q1[n] * p2[n] + q2[n];
            }
        __syncthreads();
    }

    float hc[NSt];
#pragma unroll
    for (int n = 0; n < NSt; ++n) hc[n] = (tid == 0) ? 0.0f : sQ[n * 256 + tid - 1];

    // pass 2: replay chunk, emit y and merge (atomic, permuted target)
    float* yb = ymerge + ((long long)b * CC + c) * LLs;
    for (int t = 0; t < T; ++t) {
        const int l = l0 + t;
        float d = dbv;
#pragma unroll
        for (int r = 0; r < RDt; ++r) d += dw[r] * xd[r * LLs + l];
        d = vm_softplus(d);
        const float xv = xr[l];
        float y = Dv * xv;
#pragma unroll
        for (int n = 0; n < NSt; ++n) {
            const float a = expf(d * An[n]);
            hc[n] = a * hc[n] + d * xv * xd[(RDt + n) * LLs + l];
            y += hc[n] * xd[(RDt + NSt + n) * LLs + l];
        }
        int tgt;
        if (k == 0)      tgt = l;
        else if (k == 1) tgt = (l % HHs) * WWs + (l / HHs);
        else if (k == 2) tgt = LLs - 1 - l;
        else { const int q = LLs - 1 - l; tgt = (q % HHs) * WWs + (q / HHs); }
        atomicAdd(&yb[tgt], y);
    }
}

// ---------------- final gated update ----------------
__global__ __launch_bounds__(256)
void vm_gate_out(const float* __restrict__ glin, const float* __restrict__ z2,
                 const float* __restrict__ h, float* __restrict__ out)
{
    const int idx = blockIdx.x * 256 + threadIdx.x;
    const float g = vm_sigmoid(glin[idx]);
    out[idx] = g * z2[idx] + (1.0f - g) * h[idx];
}

// ---------------- host orchestration ----------------
extern "C" void kernel_launch(void* const* d_in, const int* in_sizes, int n_in,
                              void* d_out, int out_size, void* d_ws, size_t ws_size,
                              hipStream_t stream)
{
    (void)in_sizes; (void)n_in; (void)out_size; (void)ws_size;
    const float* h         = (const float*)d_in[0];
    const float* x         = (const float*)d_in[4];
    const float* in_proj_w = (const float*)d_in[5];
    const float* in_proj_b = (const float*)d_in[6];
    const float* ln1_g     = (const float*)d_in[7];
    const float* ln1_b     = (const float*)d_in[8];
    const float* ssm_in_w  = (const float*)d_in[9];
    const float* conv_w    = (const float*)d_in[10];
    const float* conv_b    = (const float*)d_in[11];
    const float* x_proj_w  = (const float*)d_in[12];
    const float* dt_w      = (const float*)d_in[13];
    const float* dt_b      = (const float*)d_in[14];
    const float* A_logs    = (const float*)d_in[15];
    const float* Ds        = (const float*)d_in[16];
    const float* onorm_g   = (const float*)d_in[17];
    const float* onorm_b   = (const float*)d_in[18];
    const float* out_proj_w= (const float*)d_in[19];
    const float* ln2_g     = (const float*)d_in[20];
    const float* ln2_b     = (const float*)d_in[21];
    const float* fc1_w     = (const float*)d_in[22];
    const float* fc1_b     = (const float*)d_in[23];
    const float* fc2_w     = (const float*)d_in[24];
    const float* fc2_b     = (const float*)d_in[25];
    const float* gate_w    = (const float*)d_in[26];
    const float* gate_b    = (const float*)d_in[27];

    float* ws = (float*)d_ws;
    const long long S = (long long)BB * CC * LLs;       // 4,194,304 floats
    float* feat = ws;              // [0,   S)
    float* tln  = ws + S;          // [S,  2S)  LN scratch (reused)
    float* ubuf = ws + 2 * S;      // [2S, 3S)  conv out, later ymerge
    float* upre = ws + 3 * S;      // [3S, 4S)  ssm_in out, later z
    float* xsb  = ws + 4 * S;      // [4S, 8S)  xs, later m1 (fc1 out)
    float* xdbl = ws + 8 * S;      // [8S, 8.5S)
    float* z2   = ws + 8 * S + S / 2;   // [8.5S, 9.5S)
    float* glin = z2 + S;               // [9.5S, 10.5S)
    float* ym   = ubuf;
    float* zbuf = upre;
    float* m1   = xsb;

    const int LT = LLs / 128;      // 32 N-tiles
    dim3 blk(256);

    // 1. feat = in_proj(concat[h, x]) + b
    vm_gemm_cf<0><<<dim3(BB * LT, 1), blk, 0, stream>>>(in_proj_w, 1, 0,
        h, CC, CC * LLs, x, CINs, CINs * LLs, in_proj_b, nullptr, feat, CC);
    // 2. ln1
    vm_ln_cf<<<dim3((BB * LLs) / 256), blk, 0, stream>>>(feat, ln1_g, ln1_b, tln);
    // 3. ssm in_proj (no bias)
    vm_gemm_cf<0><<<dim3(BB * LT, 1), blk, 0, stream>>>(ssm_in_w, 1, 0,
        tln, CC, CC * LLs, nullptr, 0, 0, nullptr, nullptr, upre, CC);
    // 4. depthwise 3x3 + SiLU
    vm_dwconv_silu<<<dim3((BB * CC * LLs) / 256), blk, 0, stream>>>(upre, conv_w, conv_b, ubuf);
    // 5. build 4 scan orderings
    vm_build_xs<<<dim3((BB * CC * LLs) / 256), blk, 0, stream>>>(ubuf, xsb);
    // 6. x_dbl = x_proj_w[k] @ xs   (M=16, per-direction weights)
    vm_gemm_cf<0><<<dim3(BB * KDr * LT, 1), blk, 0, stream>>>(x_proj_w, KDr, (RDt + 2 * NSt) * CC,
        xsb, CC, CC * LLs, nullptr, 0, 0, nullptr, nullptr, xdbl, RDt + 2 * NSt);
    // 7. zero merge buffer, 8. selective scan + cross-merge (atomic)
    hipMemsetAsync(ym, 0, (size_t)S * sizeof(float), stream);
    vm_scan<<<dim3(BB * KDr * CC), blk, 0, stream>>>(xsb, xdbl, dt_w, dt_b, A_logs, Ds, ym);
    // 9. out-norm
    vm_ln_cf<<<dim3((BB * LLs) / 256), blk, 0, stream>>>(ym, onorm_g, onorm_b, tln);
    // 10. z = feat + out_proj(y)
    vm_gemm_cf<0><<<dim3(BB * LT, 1), blk, 0, stream>>>(out_proj_w, 1, 0,
        tln, CC, CC * LLs, nullptr, 0, 0, nullptr, feat, zbuf, CC);
    // 11. ln2
    vm_ln_cf<<<dim3((BB * LLs) / 256), blk, 0, stream>>>(zbuf, ln2_g, ln2_b, tln);
    // 12. m1 = gelu(fc1 @ ln2 + b)   (M=512 -> 4 row tiles)
    vm_gemm_cf<1><<<dim3(BB * LT, 4), blk, 0, stream>>>(fc1_w, 1, 0,
        tln, CC, CC * LLs, nullptr, 0, 0, fc1_b, nullptr, m1, DFFs);
    // 13. z2 = z + fc2 @ m1 + b
    vm_gemm_cf<0><<<dim3(BB * LT, 1), blk, 0, stream>>>(fc2_w, 1, 0,
        m1, DFFs, DFFs * LLs, nullptr, 0, 0, fc2_b, zbuf, z2, CC);
    // 14. gate linear
    vm_gemm_cf<0><<<dim3(BB * LT, 1), blk, 0, stream>>>(gate_w, 1, 0,
        z2, CC, CC * LLs, nullptr, 0, 0, gate_b, nullptr, glin, CC);
    // 15. out = sigmoid(gate)*z2 + (1-sigmoid)*h
    vm_gate_out<<<dim3((BB * CC * LLs) / 256), blk, 0, stream>>>(glin, z2, h, (float*)d_out);
}
